// SingleLayerMoE_62878321214325
// MI455X (gfx1250) — matrix operand order repaired and checked by reading, not compiled, not run on hardware
//
#include <hip/hip_runtime.h>
#include <hip/hip_bf16.h>
#include <math.h>

typedef __bf16 bf16_t;
typedef __attribute__((ext_vector_type(16))) __bf16        v16bf;
typedef __attribute__((ext_vector_type(8)))  float         v8f;
typedef __attribute__((ext_vector_type(8)))  unsigned int  v8u;
typedef __attribute__((ext_vector_type(4)))  int           v4i;

constexpr int   T     = 1024;   // tokens (B*S)
constexpr int   H     = 1024;   // hidden
constexpr int   E     = 8;      // experts
constexpr int   I     = 1024;   // intermediate
constexpr int   MT    = 64;     // token rows per GEMM block (4 x 16 WMMA subtiles)
constexpr float ALPHA = 1.702f;
constexpr float LIMIT = 7.0f;

// ---- async global->LDS staging (gfx1250 ASYNCcnt path), with safe fallback ---
#if defined(__has_builtin)
#if __has_builtin(__builtin_amdgcn_global_load_async_to_lds_b128)
#define USE_ASYNC_LDS 1
#endif
#endif
#ifndef USE_ASYNC_LDS
#define USE_ASYNC_LDS 0
#endif

#define AS1 __attribute__((address_space(1)))
#define AS3 __attribute__((address_space(3)))

__device__ __forceinline__ void stage16B(const bf16_t* g, bf16_t* lds) {
#if USE_ASYNC_LDS
  __builtin_amdgcn_global_load_async_to_lds_b128(
      (AS1 v4i*)(g), (AS3 v4i*)(lds), /*offset=*/0, /*cpol=*/0);
#else
  *reinterpret_cast<uint4*>(lds) = *reinterpret_cast<const uint4*>(g);
#endif
}

__device__ __forceinline__ void stage_wait() {
#if USE_ASYNC_LDS
#if __has_builtin(__builtin_amdgcn_s_wait_asynccnt)
  __builtin_amdgcn_s_wait_asynccnt(0);
#else
  asm volatile("s_wait_asynccnt 0" ::: "memory");
#endif
#endif
  __syncthreads();
}

// ---------------- WMMA fragment helpers (layouts per CDNA5 ISA) ----

// A (16x32 bf16): lane holds row M=lane%16; elements 0..7 = K[koff..koff+7],
// elements 8..15 = K[koff+16..koff+23], koff = 8*(lane/16). Two 16B loads.
__device__ __forceinline__ v16bf load_a_frag(const bf16_t* rowbase, int k, int lane) {
  const int koff = (lane >> 4) << 3;
  const uint4 lo = *reinterpret_cast<const uint4*>(rowbase + k + koff);
  const uint4 hi = *reinterpret_cast<const uint4*>(rowbase + k + koff + 16);
  v8u u;
  u[0] = lo.x; u[1] = lo.y; u[2] = lo.z; u[3] = lo.w;
  u[4] = hi.x; u[5] = hi.y; u[6] = hi.z; u[7] = hi.w;
  return __builtin_bit_cast(v16bf, u);
}

// B (32x16 bf16): lane holds column N=lane%16; elements 0..7 = K[koff..koff+7],
// elements 8..15 = K[koff+16..koff+23], koff = 8*(lane/16). Two 16B loads from [N][K].
__device__ __forceinline__ v16bf load_b_frag(const bf16_t* colbase, int k, int lane) {
  const bf16_t* p = colbase + k + ((lane >> 4) << 3);
  const uint4 lo = *reinterpret_cast<const uint4*>(p);
  const uint4 hi = *reinterpret_cast<const uint4*>(p + 16);
  v8u u;
  u[0] = lo.x; u[1] = lo.y; u[2] = lo.z; u[3] = lo.w;
  u[4] = hi.x; u[5] = hi.y; u[6] = hi.z; u[7] = hi.w;
  return __builtin_bit_cast(v16bf, u);
}

__device__ __forceinline__ v8f wmma_bf16(v16bf a, v16bf b, v8f c) {
  return __builtin_amdgcn_wmma_f32_16x16x32_bf16(
      /*neg_a=*/false, a, /*neg_b=*/false, b,
      /*c_mod=*/(short)0, c, /*reuse_a=*/false, /*reuse_b=*/false);
}

// ---------------- 1) zero output + expert counters ----------------------------

__global__ void moe_zero(float* __restrict__ out, int* __restrict__ counts) {
  const int idx = blockIdx.x * blockDim.x + threadIdx.x;
  if (idx < E) counts[idx] = 0;
  const int stride = gridDim.x * blockDim.x;
  for (int i = idx; i < T * H; i += stride) out[i] = 0.0f;
}

// ---------------- 2) router: softmax over E, top-2, build expert lists --------

__global__ void __launch_bounds__(256) moe_router(
    const float* __restrict__ x, const float* __restrict__ rw,
    const float* __restrict__ rb, int* __restrict__ counts,
    int* __restrict__ lists, float* __restrict__ wts) {
  __shared__ float s_rw[E * H];
  for (int i = threadIdx.x; i < E * H; i += blockDim.x) s_rw[i] = rw[i];
  __syncthreads();

  const int t = blockIdx.x * blockDim.x + threadIdx.x;
  if (t >= T) return;

  float logit[E];
#pragma unroll
  for (int e = 0; e < E; ++e) logit[e] = rb[e];
  const float* xr = x + (size_t)t * H;
  for (int h = 0; h < H; ++h) {
    const float xv = xr[h];
#pragma unroll
    for (int e = 0; e < E; ++e) logit[e] = fmaf(xv, s_rw[e * H + h], logit[e]);
  }
  float mx = logit[0];
#pragma unroll
  for (int e = 1; e < E; ++e) mx = fmaxf(mx, logit[e]);
  float p[E], sum = 0.0f;
#pragma unroll
  for (int e = 0; e < E; ++e) { p[e] = __expf(logit[e] - mx); sum += p[e]; }
  const float inv = 1.0f / sum;

  int i0 = 0; float v0 = -1.0f;
#pragma unroll
  for (int e = 0; e < E; ++e) {            // strict > : lower index wins ties
    const float sc = p[e] * inv;
    if (sc > v0) { v0 = sc; i0 = e; }
  }
  int i1 = 0; float v1 = -1.0f;
#pragma unroll
  for (int e = 0; e < E; ++e) {
    if (e == i0) continue;
    const float sc = p[e] * inv;
    if (sc > v1) { v1 = sc; i1 = e; }
  }
  const int pos0 = atomicAdd(&counts[i0], 1);
  lists[i0 * T + pos0] = t; wts[i0 * T + pos0] = v0;
  const int pos1 = atomicAdd(&counts[i1], 1);
  lists[i1 * T + pos1] = t; wts[i1 * T + pos1] = v1;
}

// ---------------- 3) fp32 -> bf16 convert / LDS-tiled weight transposes -------

__global__ void moe_cvt_x(const float* __restrict__ x, bf16_t* __restrict__ xb) {
  const int stride = gridDim.x * blockDim.x;
  for (int i = blockIdx.x * blockDim.x + threadIdx.x; i < T * H; i += stride)
    xb[i] = (bf16_t)x[i];
}

// src [E][ROWS][COLS] f32  ->  dst [E][COLS][ROWS] bf16
// 64x64 tiles staged through LDS: coalesced f32 reads AND coalesced bf16 writes.
template <int ROWS, int COLS>
__global__ void __launch_bounds__(256) moe_transpose(
    const float* __restrict__ src, bf16_t* __restrict__ dst) {
  __shared__ bf16_t tile[64][64 + 2];   // +2 bf16 pad: odd dword stride, no bank conflicts
  const int e  = blockIdx.z;
  const int r0 = blockIdx.y * 64;       // row (slow axis of src)
  const int c0 = blockIdx.x * 64;       // col (fast axis of src)
  const float* s = src + ((size_t)e * ROWS + r0) * COLS + c0;
  for (int idx = threadIdx.x; idx < 64 * 64; idx += 256) {
    const int r = idx >> 6;
    const int c = idx & 63;             // consecutive threads -> consecutive src cols
    tile[c][r] = (bf16_t)s[(size_t)r * COLS + c];
  }
  __syncthreads();
  bf16_t* d = dst + ((size_t)e * COLS + c0) * ROWS + r0;
  for (int idx = threadIdx.x; idx < 64 * 64; idx += 256) {
    const int r = idx >> 6;             // col of src = row of dst
    const int c = idx & 63;             // consecutive threads -> consecutive dst addresses
    d[(size_t)r * ROWS + c] = tile[r][c];
  }
}

// ---------------- 4) gate_up GEMM + clamped-GLU epilogue ----------------------
// grid: x = I/128 column blocks (gate & up columns paired), y = E * (T/MT)

__global__ void __launch_bounds__(256) moe_gateup(
    const bf16_t* __restrict__ xb,   // [T][H] bf16
    const bf16_t* __restrict__ guT,  // [E][2I][H] bf16 (N-major)
    const float* __restrict__ gub,   // [E][2I]
    const int* __restrict__ counts, const int* __restrict__ lists,
    bf16_t* __restrict__ act) {      // [E*T][I] bf16
  const int e  = blockIdx.y / (T / MT);
  const int m0 = (blockIdx.y % (T / MT)) * MT;
  const int ne = counts[e];
  if (m0 >= ne) return;

  __shared__ __align__(16) bf16_t sA[MT][H];   // 128 KB
  const int tid = threadIdx.x;
  for (int idx = tid; idx < MT * (H / 8); idx += 256) {
    const int row = idx / (H / 8);
    const int col = idx % (H / 8);
    int pos = m0 + row; if (pos >= ne) pos = ne - 1;   // dup last row; stores guarded
    const int tok = lists[e * T + pos];
    stage16B(xb + (size_t)tok * H + (size_t)col * 8, &sA[row][col * 8]);
  }
  stage_wait();

  const int lane = tid & 31;
  const int wave = tid >> 5;
  const int n0   = blockIdx.x * 128 + wave * 16;   // column in [0, I)
  const int nB   = n0 + (lane & 15);
  const bf16_t* bg = guT + ((size_t)e * 2 * I + nB) * H;       // gate column
  const bf16_t* bu = guT + ((size_t)e * 2 * I + I + nB) * H;   // up column

  v8f accg[MT / 16] = {};
  v8f accu[MT / 16] = {};
  const int arow = lane & 15;
  for (int k = 0; k < H; k += 32) {
    const v16bf b0 = load_b_frag(bg, k, lane);
    const v16bf b1 = load_b_frag(bu, k, lane);
#pragma unroll
    for (int s = 0; s < MT / 16; ++s) {
      const v16bf a = load_a_frag(&sA[s * 16 + arow][0], k, lane);
      accg[s] = wmma_bf16(a, b0, accg[s]);
      accu[s] = wmma_bf16(a, b1, accu[s]);
    }
  }

  const int   N     = lane & 15;
  const float biasg = gub[e * 2 * I + n0 + N];
  const float biasu = gub[e * 2 * I + I + n0 + N];
  const int   mhi   = (lane >> 4) << 3;
#pragma unroll
  for (int s = 0; s < MT / 16; ++s) {
#pragma unroll
    for (int r = 0; r < 8; ++r) {
      const int M = s * 16 + mhi + r;
      if (m0 + M >= ne) continue;
      const float g   = fminf(accg[s][r] + biasg, LIMIT);
      const float u   = fminf(fmaxf(accu[s][r] + biasu, -LIMIT), LIMIT);
      const float glu = g * (1.0f / (1.0f + __expf(-ALPHA * g)));
      act[((size_t)(e * T + m0 + M)) * I + (n0 + N)] = (bf16_t)((u + 1.0f) * glu);
    }
  }
}

// ---------------- 5) down GEMM + weighted atomic combine ----------------------
// grid: x = H/128 column blocks, y = E * (T/MT)

__global__ void __launch_bounds__(256) moe_down(
    const bf16_t* __restrict__ act,  // [E*T][I] bf16
    const bf16_t* __restrict__ dnT,  // [E][H][I] bf16 (N-major)
    const float* __restrict__ dnb,   // [E][H]
    const int* __restrict__ counts, const int* __restrict__ lists,
    const float* __restrict__ wts, float* __restrict__ out) {
  const int e  = blockIdx.y / (T / MT);
  const int m0 = (blockIdx.y % (T / MT)) * MT;
  const int ne = counts[e];
  if (m0 >= ne) return;

  __shared__ __align__(16) bf16_t sA[MT][I];   // 128 KB
  const int tid = threadIdx.x;
  for (int idx = tid; idx < MT * (I / 8); idx += 256) {
    const int row = idx / (I / 8);
    const int col = idx % (I / 8);
    stage16B(act + ((size_t)(e * T + m0 + row)) * I + (size_t)col * 8,
             &sA[row][col * 8]);
  }
  stage_wait();

  const int lane = tid & 31;
  const int wave = tid >> 5;
  const int n0   = blockIdx.x * 128 + wave * 16;   // column in [0, H)
  const int nB   = n0 + (lane & 15);
  const bf16_t* bp = dnT + ((size_t)e * H + nB) * I;

  v8f acc[MT / 16] = {};
  const int arow = lane & 15;
  for (int k = 0; k < I; k += 32) {
    const v16bf b = load_b_frag(bp, k, lane);
#pragma unroll
    for (int s = 0; s < MT / 16; ++s) {
      const v16bf a = load_a_frag(&sA[s * 16 + arow][0], k, lane);
      acc[s] = wmma_bf16(a, b, acc[s]);
    }
  }

  const int   N    = lane & 15;
  const float bias = dnb[e * H + n0 + N];
  const int   mhi  = (lane >> 4) << 3;
#pragma unroll
  for (int s = 0; s < MT / 16; ++s) {
#pragma unroll
    for (int r = 0; r < 8; ++r) {
      const int M   = s * 16 + mhi + r;
      const int pos = m0 + M;
      if (pos >= ne) continue;
      const int   tok = lists[e * T + pos];
      const float w   = wts[e * T + pos];
      atomicAdd(out + (size_t)tok * H + (n0 + N), (acc[s][r] + bias) * w);
    }
  }
}

// ---------------- host-side launch -------------------------------------------

extern "C" void kernel_launch(void* const* d_in, const int* in_sizes, int n_in,
                              void* d_out, int out_size, void* d_ws, size_t ws_size,
                              hipStream_t stream) {
  (void)in_sizes; (void)n_in; (void)out_size; (void)ws_size;
  const float* x   = (const float*)d_in[0];  // hidden_states [1,T,H]
  const float* rw  = (const float*)d_in[1];  // router_weight [E,H]
  const float* rb  = (const float*)d_in[2];  // router_bias   [E]
  const float* gu  = (const float*)d_in[3];  // gate_up_proj  [E,H,2I]
  const float* gub = (const float*)d_in[4];  // gate_up_bias  [E,2I]
  const float* dn  = (const float*)d_in[5];  // down_proj     [E,I,H]
  const float* dnb = (const float*)d_in[6];  // down_bias     [E,H]
  float* out = (float*)d_out;

  char* ws = (char*)d_ws;
  size_t off = 0;
  auto alloc = [&](size_t bytes) -> char* {
    char* p = ws + off;
    off += (bytes + 255) & ~(size_t)255;
    return p;
  };
  int*    counts = (int*)   alloc((size_t)E * sizeof(int));
  int*    lists  = (int*)   alloc((size_t)E * T * sizeof(int));
  float*  wts    = (float*) alloc((size_t)E * T * sizeof(float));
  bf16_t* xb     = (bf16_t*)alloc((size_t)T * H * sizeof(bf16_t));
  bf16_t* guT    = (bf16_t*)alloc((size_t)E * 2 * I * H * sizeof(bf16_t));
  bf16_t* dnT    = (bf16_t*)alloc((size_t)E * H * I * sizeof(bf16_t));
  bf16_t* act    = (bf16_t*)alloc((size_t)E * T * I * sizeof(bf16_t));

  moe_zero  <<<1024, 256, 0, stream>>>(out, counts);
  moe_cvt_x <<<512,  256, 0, stream>>>(x, xb);
  // gate_up_proj [E][H][2I] -> [E][2I][H]
  moe_transpose<H, 2 * I><<<dim3(2 * I / 64, H / 64, E), 256, 0, stream>>>(gu, guT);
  // down_proj    [E][I][H] -> [E][H][I]
  moe_transpose<I, H><<<dim3(H / 64, I / 64, E), 256, 0, stream>>>(dn, dnT);
  moe_router<<<T / 256, 256, 0, stream>>>(x, rw, rb, counts, lists, wts);
  moe_gateup<<<dim3(I / 128, E * (T / MT)), 256, 0, stream>>>(xb, guT, gub, counts, lists, act);
  moe_down  <<<dim3(H / 128, E * (T / MT)), 256, 0, stream>>>(act, dnT, dnb, counts, lists, wts, out);
}
